// MultiLayerRangeAttentionNoD_8108898255319
// MI455X (gfx1250) — matrix operand order repaired and checked by reading, not compiled, not run on hardware
//
#include <hip/hip_runtime.h>
#include <hip/hip_bf16.h>

typedef __attribute__((ext_vector_type(16))) _Float16 v16h;
typedef __attribute__((ext_vector_type(8)))  _Float16 v8h;
typedef __attribute__((ext_vector_type(8)))  float    v8f;

#define NRANGES 700
#define KPTS    250
#define KPAD    256
#define IN_DIM  7
#define HID     32
#define NLAYERS 2
// (1/sqrt(HID)) * log2(e): softmax done in base-2 so v_exp_f32 needs no pre-multiply
#define SCALE_LOG2E 0.25503485871331394f
#define NEG_BIG     -3.0e38f

__global__ __launch_bounds__(256)
void range_attn_kernel(const float* __restrict__ power,
                       const int*   __restrict__ ele_idx,
                       const int*   __restrict__ azi_idx,
                       const float* __restrict__ ele_emb,
                       const float* __restrict__ azi_emb,
                       const float* __restrict__ Wq, const float* __restrict__ bq,
                       const float* __restrict__ Wk, const float* __restrict__ bk,
                       const float* __restrict__ Wv, const float* __restrict__ bv,
                       float* __restrict__ out)
{
    // f16 Q/K tiles laid out row-major [row][feature] so WMMA fragments are b128 LDS loads
    __shared__ __align__(64) _Float16 qh[KPAD][HID];   // 16 KB (q pre-scaled by SCALE_LOG2E)
    __shared__ __align__(64) _Float16 kh[KPAD][HID];   // 16 KB
    __shared__ float vsh[KPAD];                        // per-key value
    __shared__ float xsh[KPAD];                        // layer output / next-layer input
    __shared__ float wq_s[IN_DIM * HID], wk_s[IN_DIM * HID], wv_s[IN_DIM];
    __shared__ float bq_s[HID], bk_s[HID];
    __shared__ float bv_s;

    const int r    = blockIdx.x;
    const int t    = threadIdx.x;      // 0..255, row id in phase 1
    const int lane = t & 31;
    const int wid  = t >> 5;           // wave 0..7
    const int lh   = lane >> 4;        // lane half (0/1)
    const int n    = lane & 15;        // column-within-tile / A-row-within-tile

    // Per-row static features (embeddings + power); padded rows are zero.
    float feat[IN_DIM];
    if (t < KPTS) {
        const int gi = r * KPTS + t;
        feat[0] = power[gi];
        const int ei = ele_idx[gi];
        const int ai = azi_idx[gi];
        feat[1] = ele_emb[ei * 3 + 0]; feat[2] = ele_emb[ei * 3 + 1]; feat[3] = ele_emb[ei * 3 + 2];
        feat[4] = azi_emb[ai * 3 + 0]; feat[5] = azi_emb[ai * 3 + 1]; feat[6] = azi_emb[ai * 3 + 2];
    } else {
        #pragma unroll
        for (int d = 0; d < IN_DIM; ++d) feat[d] = 0.f;
    }

    for (int l = 0; l < NLAYERS; ++l) {
        __syncthreads();   // prev phase-2 (reads qh/kh, writes xsh) vs. rewrite below

        // ---- stage layer weights into LDS ----
        if (t < IN_DIM * HID) {
            wq_s[t] = Wq[l * IN_DIM * HID + t];
            wk_s[t] = Wk[l * IN_DIM * HID + t];
        }
        if (t < HID)    { bq_s[t] = bq[l * HID + t]; bk_s[t] = bk[l * HID + t]; }
        if (t < IN_DIM) { wv_s[t] = Wv[l * IN_DIM + t]; }
        if (t == 0)     { bv_s = bv[l]; }
        if (l > 0) feat[0] = (t < KPTS) ? xsh[t] : 0.f;
        __syncthreads();

        // ---- phase 1: q/k/v projections (7 -> 32 / 1); q pre-scaled; f16 into LDS ----
        if (t < KPTS) {
            #pragma unroll
            for (int j = 0; j < HID; ++j) {
                float aq = bq_s[j], ak = bk_s[j];
                #pragma unroll
                for (int d = 0; d < IN_DIM; ++d) {
                    aq += feat[d] * wq_s[d * HID + j];
                    ak += feat[d] * wk_s[d * HID + j];
                }
                qh[t][j] = (_Float16)(aq * SCALE_LOG2E);   // fold scale + log2e into q
                kh[t][j] = (_Float16)ak;
            }
            float vv = bv_s;
            #pragma unroll
            for (int d = 0; d < IN_DIM; ++d) vv += feat[d] * wv_s[d];
            vsh[t] = vv;
        } else {
            #pragma unroll
            for (int j = 0; j < HID; ++j) { qh[t][j] = (_Float16)0.f; kh[t][j] = (_Float16)0.f; }
            vsh[t] = 0.f;
        }
        __syncthreads();

        // ---- phase 2: scores via WMMA + online (2-chunk) softmax + probs@v ----
        // 16 row-tiles of 16 rows; each of 8 waves owns 2 row-tiles.
        #pragma unroll 1
        for (int mtp = 0; mtp < 2; ++mtp) {
            const int mt   = wid + mtp * 8;
            const int arow = mt * 16 + n;                 // A: M = lane%16

            // A fragment: lane holds row `arow`, K pairs per ISA 16-bit A 16x32 layout
            v8h alo = *(const v8h*)&qh[arow][lh * 8];          // K = lh*8 .. lh*8+7
            v8h ahi = *(const v8h*)&qh[arow][16 + lh * 8];     // K = 16+lh*8 ..
            v16h A;
            #pragma unroll
            for (int i = 0; i < 8; ++i) { A[i] = alo[i]; A[8 + i] = ahi[i]; }

            // online softmax running state per owned row (row = mt*16 + lh*8 + i)
            float mrun[8], srun[8], svrun[8];
            #pragma unroll
            for (int i = 0; i < 8; ++i) { mrun[i] = NEG_BIG; srun[i] = 0.f; svrun[i] = 0.f; }

            #pragma unroll 1
            for (int h2 = 0; h2 < 2; ++h2) {
                v8f zero = {};
                v8f acc[8];
                #pragma unroll
                for (int c = 0; c < 8; ++c) acc[c] = zero;

                // 8 WMMAs into this chunk's accumulators
                #pragma unroll
                for (int c = 0; c < 8; ++c) {
                    const int key = (h2 * 8 + c) * 16 + n;     // B: N = lane%16
                    v8h blo = *(const v8h*)&kh[key][lh * 8];
                    v8h bhi = *(const v8h*)&kh[key][16 + lh * 8];
                    v16h B;
                    #pragma unroll
                    for (int i = 0; i < 8; ++i) { B[i] = blo[i]; B[8 + i] = bhi[i]; }
                    acc[c] = __builtin_amdgcn_wmma_f32_16x16x32_f16(
                        false, A, false, B, (short)0, acc[c], false, false);
                }

                // per-lane value column cache
                float vcol[8];
                #pragma unroll
                for (int c = 0; c < 8; ++c) vcol[c] = vsh[(h2 * 8 + c) * 16 + n];

                // mask padded keys: only the last col-tile of chunk 1 can touch col >= 250
                const float mb = ((h2 * 8 + 7) * 16 + n >= KPTS) ? NEG_BIG : 0.f;
                #pragma unroll
                for (int i = 0; i < 8; ++i) acc[7][i] += mb;

                // online merge for each of the 8 rows this lane-half owns
                #pragma unroll
                for (int i = 0; i < 8; ++i) {
                    float vals[8];
                    float cm = NEG_BIG;
                    #pragma unroll
                    for (int c = 0; c < 8; ++c) {
                        float s = acc[c][i];
                        vals[c] = s;
                        cm = fmaxf(cm, s);
                    }
                    #pragma unroll
                    for (int mask = 1; mask < 16; mask <<= 1)
                        cm = fmaxf(cm, __shfl_xor(cm, mask, 32));   // reduce over 16-lane half

                    const float newm    = fmaxf(mrun[i], cm);
                    const float rescale = __builtin_exp2f(mrun[i] - newm);

                    float sum = 0.f, sv = 0.f;
                    #pragma unroll
                    for (int c = 0; c < 8; ++c) {
                        float p = __builtin_exp2f(vals[c] - newm);  // base-2 softmax
                        sum += p;
                        sv  += p * vcol[c];
                    }
                    #pragma unroll
                    for (int mask = 1; mask < 16; mask <<= 1) {
                        sum += __shfl_xor(sum, mask, 32);
                        sv  += __shfl_xor(sv,  mask, 32);
                    }

                    srun[i]  = srun[i]  * rescale + sum;
                    svrun[i] = svrun[i] * rescale + sv;
                    mrun[i]  = newm;
                }
            }

            // write row outputs (one lane per row)
            #pragma unroll
            for (int i = 0; i < 8; ++i) {
                if (n == 0) {
                    const int row = mt * 16 + lh * 8 + i;
                    const float o = svrun[i] / srun[i];
                    xsh[row] = o;
                    if (l == NLAYERS - 1 && row < KPTS) out[r * KPTS + row] = o;
                }
            }
        }
    }
}

extern "C" void kernel_launch(void* const* d_in, const int* in_sizes, int n_in,
                              void* d_out, int out_size, void* d_ws, size_t ws_size,
                              hipStream_t stream) {
    const float* power   = (const float*)d_in[0];
    const int*   ele     = (const int*)  d_in[1];
    const int*   azi     = (const int*)  d_in[2];
    const float* ele_emb = (const float*)d_in[3];
    const float* azi_emb = (const float*)d_in[4];
    const float* Wq      = (const float*)d_in[5];
    const float* bq      = (const float*)d_in[6];
    const float* Wk      = (const float*)d_in[7];
    const float* bk      = (const float*)d_in[8];
    const float* Wv      = (const float*)d_in[9];
    const float* bv      = (const float*)d_in[10];

    range_attn_kernel<<<NRANGES, 256, 0, stream>>>(
        power, ele, azi, ele_emb, azi_emb, Wq, bq, Wk, bk, Wv, bv, (float*)d_out);
}
